// MindBlock_73521250173373
// MI455X (gfx1250) — matrix operand order repaired and checked by reading, not compiled, not run on hardware
//
#include <hip/hip_runtime.h>

// ---------- types ----------
typedef unsigned short bf_t;
typedef unsigned int   u32;
typedef unsigned long long u64;
typedef __bf16 v16bf __attribute__((ext_vector_type(16)));
typedef float  v8f   __attribute__((ext_vector_type(8)));

union Frag { uint4 u[2]; v16bf v; };

// ---------- bf16 helpers ----------
__device__ __forceinline__ bf_t f2bf(float f) {
    union { float f; u32 u; } v; v.f = f;
    u32 u = v.u;
    u32 r = (u + 0x7fffu + ((u >> 16) & 1u)) >> 16;
    return (bf_t)r;
}
__device__ __forceinline__ float bf2f(bf_t h) {
    union { u32 u; float f; } v; v.u = ((u32)h) << 16; return v.f;
}
__device__ __forceinline__ float bflo(u32 u) {
    union { u32 x; float f; } v; v.x = u << 16; return v.f;
}
__device__ __forceinline__ float bfhi(u32 u) {
    union { u32 x; float f; } v; v.x = u & 0xffff0000u; return v.f;
}
__device__ __forceinline__ u32 pack2(float lo, float hi) {
    return (u32)f2bf(lo) | ((u32)f2bf(hi) << 16);
}

// ---------- CDNA5 async global -> LDS copy (ASYNCcnt-tracked) ----------
__device__ __forceinline__ void async_load_b128(u32 lds_off, const void* gptr) {
    asm volatile("global_load_async_to_lds_b128 %0, %1, off"
                 :: "v"(lds_off), "v"((u64)(size_t)gptr) : "memory");
}
__device__ __forceinline__ void wait_asynccnt0() {
    asm volatile("s_wait_asynccnt 0x0" ::: "memory");
}

// ---------- tiny utility kernels ----------
__global__ void zero_floats(float* p, int n) {
    int i = blockIdx.x * blockDim.x + threadIdx.x;
    if (i < n) p[i] = 0.0f;
}

__global__ void cvt_f32_bf16(const float* __restrict__ in, bf_t* __restrict__ out, int n) {
    int i = blockIdx.x * blockDim.x + threadIdx.x;
    int stride = gridDim.x * blockDim.x;
    for (; i < n; i += stride) out[i] = f2bf(in[i]);
}

// r = q + 0.1*k  (bf16 in/out)
__global__ void mix_qk(const bf_t* __restrict__ q, const bf_t* __restrict__ k,
                       bf_t* __restrict__ r, int n) {
    int i = blockIdx.x * blockDim.x + threadIdx.x;
    int stride = gridDim.x * blockDim.x;
    for (; i < n; i += stride)
        r[i] = f2bf(bf2f(q[i]) + 0.1f * bf2f(k[i]));
}

// ---------- LayerNorm (block-per-token, D = 2048 = 256*8) ----------
__global__ __launch_bounds__(256) void layernorm_bf16(
    const float* __restrict__ x, const float* __restrict__ g,
    const float* __restrict__ be, bf_t* __restrict__ h, int D)
{
    __shared__ float r1[256];
    __shared__ float r2[256];
    const int t = blockIdx.x, tid = threadIdx.x, d0 = tid * 8;
    const float* xr = x + (size_t)t * D;
    float4 a = ((const float4*)(xr + d0))[0];
    float4 b = ((const float4*)(xr + d0))[1];
    float s1 = a.x + a.y + a.z + a.w + b.x + b.y + b.z + b.w;
    float s2 = a.x*a.x + a.y*a.y + a.z*a.z + a.w*a.w
             + b.x*b.x + b.y*b.y + b.z*b.z + b.w*b.w;
    r1[tid] = s1; r2[tid] = s2; __syncthreads();
    for (int off = 128; off; off >>= 1) {
        if (tid < off) { r1[tid] += r1[tid + off]; r2[tid] += r2[tid + off]; }
        __syncthreads();
    }
    const float mean = r1[0] * (1.0f / D);
    const float var  = r2[0] * (1.0f / D) - mean * mean;
    const float rstd = rsqrtf(var + 1e-5f);
    float4 g0 = ((const float4*)(g + d0))[0];
    float4 g1 = ((const float4*)(g + d0))[1];
    float4 b0 = ((const float4*)(be + d0))[0];
    float4 b1 = ((const float4*)(be + d0))[1];
    float o0 = (a.x - mean) * rstd * g0.x + b0.x;
    float o1 = (a.y - mean) * rstd * g0.y + b0.y;
    float o2 = (a.z - mean) * rstd * g0.z + b0.z;
    float o3 = (a.w - mean) * rstd * g0.w + b0.w;
    float o4 = (b.x - mean) * rstd * g1.x + b1.x;
    float o5 = (b.y - mean) * rstd * g1.y + b1.y;
    float o6 = (b.z - mean) * rstd * g1.z + b1.z;
    float o7 = (b.w - mean) * rstd * g1.w + b1.w;
    uint4 o;
    o.x = pack2(o0, o1); o.y = pack2(o2, o3);
    o.z = pack2(o4, o5); o.w = pack2(o6, o7);
    *(uint4*)(h + (size_t)t * D + d0) = o;
}

// ---------- WMMA bf16 GEMM:  out[M,N] = A[M,K] @ W[N,K]^T ----------
// Double-buffered LDS, async global->LDS copies, 8 waves:
//   block tile BM x BN_, wave tile 32 x (BN_/2), K step 32.
// Epilogue is compile-time configured -> straight-line stores (no per-element
// branches / guarded loads).
#define BM 128
#define BK 32
#define BKP 40   // padded LDS row (ushorts)

template <int BN_, bool HAS_BIAS, bool HAS_ROWSCALE, bool HAS_RESID, bool OUT_BF>
__global__ __launch_bounds__(256) void gemm_bf16_wmma(
    const bf_t* __restrict__ A, const bf_t* __restrict__ W,
    bf_t* __restrict__ outB, float* __restrict__ outF,
    int M, int N, int K,
    const float* __restrict__ bias, const float* __restrict__ rowscale,
    const float* __restrict__ residual, const float* __restrict__ scalarp,
    float scalarmul)
{
    constexpr int NT  = BN_ / 32;               // N subtiles (16 wide) per wave
    constexpr int NCB = (BN_ * BK) / (256 * 8); // b128 chunks/thread for B tile

    __shared__ bf_t As[2][BM][BKP];
    __shared__ bf_t Bs[2][BN_][BKP];

    const int tid  = threadIdx.x;
    const int lane = tid & 31;
    const int wave = tid >> 5;
    const int wm = wave & 3;     // 32-row band
    const int wn = wave >> 2;    // (BN_/2)-col band
    const int m0blk = blockIdx.y * BM;
    const int n0blk = blockIdx.x * BN_;

    const int halo = lane >> 4;   // 0|1
    const int l15  = lane & 15;
    const int akb  = halo * 8;    // A frag K base (ISA 16-bit A 16x32 layout)
    const int bkb  = halo * 8;    // B frag K base

    const v8f vzero = {0.f, 0.f, 0.f, 0.f, 0.f, 0.f, 0.f, 0.f};
    v8f acc[2][NT];
#pragma unroll
    for (int i = 0; i < 2; ++i)
#pragma unroll
        for (int j = 0; j < NT; ++j) acc[i][j] = vzero;

    // async copy of one K-tile into LDS buffer `buf`
    auto issue_tile = [&](int k0, int buf) {
#pragma unroll
        for (int i = 0; i < 2; ++i) {                 // A: 512 x b128 chunks
            int c = tid + i * 256;
            int r = c >> 2, off = (c & 3) * 8;
            async_load_b128((u32)(size_t)&As[buf][r][off],
                            A + (size_t)(m0blk + r) * K + k0 + off);
        }
#pragma unroll
        for (int i = 0; i < NCB; ++i) {               // B: BN_*4 x b128 chunks
            int c = tid + i * 256;
            int r = c >> 2, off = (c & 3) * 8;
            async_load_b128((u32)(size_t)&Bs[buf][r][off],
                            W + (size_t)(n0blk + r) * K + k0 + off);
        }
    };

    const int nsteps = K / BK;
    issue_tile(0, 0);
    wait_asynccnt0();
    __syncthreads();

    for (int s = 0; s < nsteps; ++s) {
        const int cur = s & 1;
        const int nxt = cur ^ 1;
        if (s + 1 < nsteps) issue_tile((s + 1) * BK, nxt);

        Frag aF[2], bF[NT];
#pragma unroll
        for (int mi = 0; mi < 2; ++mi) {
            const int m = wm * 32 + mi * 16 + l15;
            aF[mi].u[0] = *(const uint4*)(&As[cur][m][akb]);
            aF[mi].u[1] = *(const uint4*)(&As[cur][m][akb + 16]);
        }
#pragma unroll
        for (int ni = 0; ni < NT; ++ni) {
            const int n = wn * (BN_ / 2) + ni * 16 + l15;
            const uint4* p = (const uint4*)(&Bs[cur][n][bkb]);
            bF[ni].u[0] = p[0]; bF[ni].u[1] = p[2];
        }
#pragma unroll
        for (int mi = 0; mi < 2; ++mi)
#pragma unroll
            for (int ni = 0; ni < NT; ++ni)
                acc[mi][ni] = __builtin_amdgcn_wmma_f32_16x16x32_bf16(
                    false, aF[mi].v, false, bF[ni].v, (short)0, acc[mi][ni],
                    false, false);

        if (s + 1 < nsteps) wait_asynccnt0();
        __syncthreads();   // also waits DScnt -> safe to overwrite `cur` next iter
    }

    float sadd = 0.0f;
    if (HAS_RESID) sadd = (*scalarp) * scalarmul;
#pragma unroll
    for (int mi = 0; mi < 2; ++mi) {
#pragma unroll
        for (int ni = 0; ni < NT; ++ni) {
            const int nglob = n0blk + wn * (BN_ / 2) + ni * 16 + l15;
            float bv = 0.0f;
            if (HAS_BIAS) bv = bias[nglob];
#pragma unroll
            for (int r = 0; r < 8; ++r) {
                const int mglob = m0blk + wm * 32 + mi * 16 + r + 8 * halo;
                float v = acc[mi][ni][r] + bv;
                if (HAS_ROWSCALE) v *= rowscale[mglob];
                if (HAS_RESID)    v += sadd + residual[(size_t)mglob * N + nglob];
                if (OUT_BF) outB[(size_t)mglob * N + nglob] = f2bf(v);
                else        outF[(size_t)mglob * N + nglob] = v;
            }
        }
    }
}

// ---------- router: softmax/argmax/top-1 weight + norm reg (wave-per-token) ----------
__global__ __launch_bounds__(256) void router_kernel(
    const bf_t* __restrict__ qb, const bf_t* __restrict__ kb,
    const float* __restrict__ logits,
    int* __restrict__ cid, float* __restrict__ wtok,
    float* __restrict__ regAcc, int T, int D)
{
    const int lane = threadIdx.x & 31;
    const int wave = threadIdx.x >> 5;
    const int t = blockIdx.x * 8 + wave;
    if (t >= T) return;
    const bf_t* q = qb + (size_t)t * D;
    const bf_t* k = kb + (size_t)t * D;
    float qs = 0.f, ks = 0.f;
    for (int d = lane * 4; d < D; d += 128) {
        uint2 a = *(const uint2*)(q + d);
        uint2 b = *(const uint2*)(k + d);
        float q0 = bflo(a.x), q1 = bfhi(a.x), q2 = bflo(a.y), q3 = bfhi(a.y);
        float k0 = bflo(b.x), k1 = bfhi(b.x), k2 = bflo(b.y), k3 = bfhi(b.y);
        qs += q0*q0 + q1*q1 + q2*q2 + q3*q3;
        ks += k0*k0 + k1*k1 + k2*k2 + k3*k3;
    }
    for (int off = 16; off; off >>= 1) {
        qs += __shfl_xor(qs, off, 32);
        ks += __shfl_xor(ks, off, 32);
    }
    const float* L = logits + (size_t)t * 64;
    float l0 = L[lane], l1 = L[lane + 32];
    float m; int mi;
    if (l0 >= l1) { m = l0; mi = lane; } else { m = l1; mi = lane + 32; }
    for (int off = 16; off; off >>= 1) {
        float om = __shfl_xor(m, off, 32);
        int   oi = __shfl_xor(mi, off, 32);
        if (om > m || (om == m && oi < mi)) { m = om; mi = oi; }
    }
    float e = __expf(l0 - m) + __expf(l1 - m);
    for (int off = 16; off; off >>= 1) e += __shfl_xor(e, off, 32);
    if (lane == 0) {
        cid[t]  = mi;
        wtok[t] = 1.0f / e;   // top-1 softmax weight
        atomicAdd(regAcc, sqrtf(qs) + sqrtf(ks));
    }
}

// ---------- channel scatter-mean: block per (c,b) ----------
__global__ __launch_bounds__(256) void channel_agg(
    const int* __restrict__ cid, const float* __restrict__ wtok,
    const bf_t* __restrict__ vb, bf_t* __restrict__ chm,
    int S, int D, int C)
{
    const int c = blockIdx.x, b = blockIdx.y;
    const int base = b * S;
    const int d0 = threadIdx.x * 8;
    float acc[8];
#pragma unroll
    for (int i = 0; i < 8; ++i) acc[i] = 0.f;
    float den = 0.f;
    for (int s = 0; s < S; ++s) {
        if (cid[base + s] == c) {
            float ww = wtok[base + s];
            den += ww;
            uint4 raw = *(const uint4*)(vb + (size_t)(base + s) * D + d0);
            acc[0] += ww * bflo(raw.x); acc[1] += ww * bfhi(raw.x);
            acc[2] += ww * bflo(raw.y); acc[3] += ww * bfhi(raw.y);
            acc[4] += ww * bflo(raw.z); acc[5] += ww * bfhi(raw.z);
            acc[6] += ww * bflo(raw.w); acc[7] += ww * bfhi(raw.w);
        }
    }
    float inv = 1.f / (den + 1e-8f);
    uint4 o;
    o.x = pack2(acc[0] * inv, acc[1] * inv);
    o.y = pack2(acc[2] * inv, acc[3] * inv);
    o.z = pack2(acc[4] * inv, acc[5] * inv);
    o.w = pack2(acc[6] * inv, acc[7] * inv);
    *(uint4*)(chm + (size_t)(b * C + c) * D + d0) = o;
}

// ---------- gather channel summary back to tokens, scale by w ----------
__global__ __launch_bounds__(256) void gather_scale(
    const int* __restrict__ cid, const float* __restrict__ wtok,
    const float* __restrict__ cho, bf_t* __restrict__ agg,
    int S, int D, int C)
{
    const int t = blockIdx.x;
    const int b = t / S;
    const int c = cid[t];
    const float ww = wtok[t];
    const float* co = cho + (size_t)(b * C + c) * D;
    const int d0 = threadIdx.x * 8;
    float4 a  = ((const float4*)(co + d0))[0];
    float4 b4 = ((const float4*)(co + d0))[1];
    uint4 o;
    o.x = pack2(a.x * ww, a.y * ww);
    o.y = pack2(a.z * ww, a.w * ww);
    o.z = pack2(b4.x * ww, b4.y * ww);
    o.w = pack2(b4.z * ww, b4.w * ww);
    *(uint4*)(agg + (size_t)t * D + d0) = o;
}

// ---------- host launcher ----------
extern "C" void kernel_launch(void* const* d_in, const int* in_sizes, int n_in,
                              void* d_out, int out_size, void* d_ws, size_t ws_size,
                              hipStream_t stream) {
    (void)in_sizes; (void)n_in; (void)out_size; (void)ws_size;
    const float* x    = (const float*)d_in[0];
    const float* mask = (const float*)d_in[1];
    const float* g    = (const float*)d_in[2];
    const float* be   = (const float*)d_in[3];
    const float* Wq   = (const float*)d_in[4];
    const float* Wk   = (const float*)d_in[5];
    const float* Wv   = (const float*)d_in[6];
    const float* Wr   = (const float*)d_in[7];
    const float* br   = (const float*)d_in[8];
    const float* Wa   = (const float*)d_in[9];
    const float* ba   = (const float*)d_in[10];
    const float* Wout = (const float*)d_in[11];
    const float* bout = (const float*)d_in[12];
    float* out = (float*)d_out;

    const int B = 4, S = 4096, D = 2048, C = 64;
    const int T = B * S;

    char* ws = (char*)d_ws;
    auto alloc = [&](size_t bytes) -> char* {
        char* p = ws; ws += (bytes + 255) & ~(size_t)255; return p;
    };
    bf_t* hbuf = (bf_t*)alloc((size_t)T * D * 2);   // h, then r = q + 0.1k
    bf_t* qbuf = (bf_t*)alloc((size_t)T * D * 2);   // q, then agg
    bf_t* kbuf = (bf_t*)alloc((size_t)T * D * 2);
    bf_t* vbuf = (bf_t*)alloc((size_t)T * D * 2);
    bf_t* wqb  = (bf_t*)alloc((size_t)D * D * 2);
    bf_t* wkb  = (bf_t*)alloc((size_t)D * D * 2);
    bf_t* wvb  = (bf_t*)alloc((size_t)D * D * 2);
    bf_t* wab  = (bf_t*)alloc((size_t)D * D * 2);
    bf_t* wob  = (bf_t*)alloc((size_t)D * D * 2);
    bf_t* wrb  = (bf_t*)alloc((size_t)C * D * 2);
    float* logits = (float*)alloc((size_t)T * C * 4);
    int*   cid    = (int*)alloc((size_t)T * 4);
    float* wtok   = (float*)alloc((size_t)T * 4);
    bf_t*  chm    = (bf_t*)alloc((size_t)B * C * D * 2);
    float* cho    = (float*)alloc((size_t)B * C * D * 4);
    float* reg    = (float*)alloc(256);

    // 0) zero the reg accumulator (workspace is poisoned, not re-zeroed)
    zero_floats<<<1, 64, 0, stream>>>(reg, 1);

    // 1) weight conversion f32 -> bf16
    cvt_f32_bf16<<<4096, 256, 0, stream>>>(Wq,   wqb, D * D);
    cvt_f32_bf16<<<4096, 256, 0, stream>>>(Wk,   wkb, D * D);
    cvt_f32_bf16<<<4096, 256, 0, stream>>>(Wv,   wvb, D * D);
    cvt_f32_bf16<<<4096, 256, 0, stream>>>(Wa,   wab, D * D);
    cvt_f32_bf16<<<4096, 256, 0, stream>>>(Wout, wob, D * D);
    cvt_f32_bf16<<<512,  256, 0, stream>>>(Wr,   wrb, C * D);

    // 2) LayerNorm -> h (bf16)
    layernorm_bf16<<<T, 256, 0, stream>>>(x, g, be, hbuf, D);

    // 3) q/k/v GEMMs: outB = bf16, rowscale = mask
    dim3 blk(256);
    dim3 gBig(D / 128, T / BM);   // (16, 128)
    gemm_bf16_wmma<128, false, true, false, true><<<gBig, blk, 0, stream>>>(
        hbuf, wqb, qbuf, nullptr, T, D, D, nullptr, mask, nullptr, nullptr, 0.f);
    gemm_bf16_wmma<128, false, true, false, true><<<gBig, blk, 0, stream>>>(
        hbuf, wkb, kbuf, nullptr, T, D, D, nullptr, mask, nullptr, nullptr, 0.f);
    gemm_bf16_wmma<128, false, true, false, true><<<gBig, blk, 0, stream>>>(
        hbuf, wvb, vbuf, nullptr, T, D, D, nullptr, mask, nullptr, nullptr, 0.f);

    // 4) routing input r = q + 0.1k (reuse h buffer)
    mix_qk<<<8192, 256, 0, stream>>>(qbuf, kbuf, hbuf, T * D);

    // 5) router logits: [T,64] = r @ Wr^T + br   (narrow-N variant, f32 out)
    dim3 gLog(C / 64, T / BM);   // (1, 128)
    gemm_bf16_wmma<64, true, false, false, false><<<gLog, blk, 0, stream>>>(
        hbuf, wrb, nullptr, logits, T, C, D, br, nullptr, nullptr, nullptr, 0.f);

    // 6) softmax/argmax/top-1 weight + reg accumulation
    router_kernel<<<T / 8, 256, 0, stream>>>(qbuf, kbuf, logits, cid, wtok, reg, T, D);

    // 7) channel scatter-mean -> ch_mean (bf16)
    channel_agg<<<dim3(C, B), blk, 0, stream>>>(cid, wtok, vbuf, chm, S, D, C);

    // 8) ch_out = ch_mean @ Wa^T + ba   (M = B*C = 256, f32 out)
    dim3 gCh(D / 128, (B * C) / BM);  // (16, 2)
    gemm_bf16_wmma<128, true, false, false, false><<<gCh, blk, 0, stream>>>(
        chm, wab, nullptr, cho, B * C, D, D, ba, nullptr, nullptr, nullptr, 0.f);

    // 9) gather ch_out back to tokens * w  -> agg (reuse q buffer)
    gather_scale<<<T, 256, 0, stream>>>(cid, wtok, cho, qbuf, S, D, C);

    // 10) output = agg @ Wout^T + bout + reg*(0.001/T) + residual(x)
    gemm_bf16_wmma<128, true, false, true, false><<<gBig, blk, 0, stream>>>(
        qbuf, wob, nullptr, out, T, D, D, bout, nullptr, x, reg, 0.001f / (float)T);
}
